// NostARHeadAttention_88794153878119
// MI455X (gfx1250) — matrix operand order repaired and checked by reading, not compiled, not run on hardware
//
#include <hip/hip_runtime.h>

#define E_ 2048
#define S_ 2048
#define B_ 8
#define H_ 16
#define D_ 128
#define PAD_TOK 50257

typedef __attribute__((ext_vector_type(16))) __bf16 v16bf;
typedef __attribute__((ext_vector_type(8)))  float  v8f;
typedef __attribute__((ext_vector_type(4)))  unsigned int v4u;
typedef __attribute__((ext_vector_type(8)))  int    v8i;
typedef __attribute__((ext_vector_type(4)))  int    v4i;

union FragU { uint4 q[2]; v16bf v; };

__device__ __forceinline__ unsigned short f2bf(float f) {
  unsigned int u = __float_as_uint(f);
  u += 0x7FFFu + ((u >> 16) & 1u);   // round-to-nearest-even
  return (unsigned short)(u >> 16);
}
__device__ __forceinline__ float bf2f(unsigned short h) {
  return __uint_as_float((unsigned int)h << 16);
}
__device__ __forceinline__ unsigned pack2(float a, float b) {
  return (unsigned)f2bf(a) | ((unsigned)f2bf(b) << 16);
}

// ---------------------------------------------------------------- kernel 0
// ix[b] = last non-pad index (== argmax of cumsum of nonpad); all-pad -> 0.
__global__ void k_find_ix(const int* __restrict__ ids, int* __restrict__ ix) {
  int b = blockIdx.x, tid = threadIdx.x;
  __shared__ int red[256];
  int best = -1;
  for (int s = tid; s < S_; s += 256)
    if (ids[b * S_ + s] != PAD_TOK) best = s;
  red[tid] = best;
  __syncthreads();
  for (int off = 128; off > 0; off >>= 1) {
    if (tid < off) red[tid] = max(red[tid], red[tid + off]);
    __syncthreads();
  }
  if (tid == 0) ix[b] = red[0] < 0 ? 0 : red[0];
}

// ---------------------------------------------------------------- kernel 1
// Fused LayerNorm -> bf16: one block per row; stats in LDS, then each thread
// normalizes and packs its 8 elements into a single 16-byte store.
__global__ void k_ln_bf16(const float* __restrict__ x, const float* __restrict__ lnw,
                          const float* __restrict__ lnb, unsigned short* __restrict__ out) {
  int r = blockIdx.x, tid = threadIdx.x;
  const float4* xp = (const float4*)(x + (size_t)r * E_);
  float4 a = xp[tid * 2], b = xp[tid * 2 + 1];
  float s = a.x + a.y + a.z + a.w + b.x + b.y + b.z + b.w;
  float q = a.x * a.x + a.y * a.y + a.z * a.z + a.w * a.w +
            b.x * b.x + b.y * b.y + b.z * b.z + b.w * b.w;
  __shared__ float rs[256], rq[256];
  __shared__ float2 st;
  rs[tid] = s; rq[tid] = q;
  __syncthreads();
  for (int off = 128; off > 0; off >>= 1) {
    if (tid < off) { rs[tid] += rs[tid + off]; rq[tid] += rq[tid + off]; }
    __syncthreads();
  }
  if (tid == 0) {
    float mu  = rs[0] * (1.0f / E_);
    float var = rq[0] * (1.0f / E_) - mu * mu;
    st = make_float2(mu, rsqrtf(var + 1e-5f));
  }
  __syncthreads();
  float mu = st.x, rsig = st.y;
  int c = tid * 8;
  float4 w0 = *(const float4*)(lnw + c), w1 = *(const float4*)(lnw + c + 4);
  float4 b0 = *(const float4*)(lnb + c), b1 = *(const float4*)(lnb + c + 4);
  uint4 o;
  o.x = pack2((a.x - mu) * rsig * w0.x + b0.x, (a.y - mu) * rsig * w0.y + b0.y);
  o.y = pack2((a.z - mu) * rsig * w0.z + b0.z, (a.w - mu) * rsig * w0.w + b0.w);
  o.z = pack2((b.x - mu) * rsig * w1.x + b1.x, (b.y - mu) * rsig * w1.y + b1.y);
  o.w = pack2((b.z - mu) * rsig * w1.z + b1.z, (b.w - mu) * rsig * w1.w + b1.w);
  *(uint4*)(out + (size_t)r * E_ + c) = o;
}

// ---------------------------------------------------------------- kernel 2
// fp32 -> bf16 weight conversion (k_w and v_w), 8 elements/thread.
__global__ void k_w2bf(const float* __restrict__ kw, const float* __restrict__ vw,
                       unsigned short* __restrict__ kwb, unsigned short* __restrict__ vwb) {
  size_t i = ((size_t)blockIdx.x * 256 + threadIdx.x) * 8;
  float4 a = *(const float4*)(kw + i), b = *(const float4*)(kw + i + 4);
  uint4 o;
  o.x = pack2(a.x, a.y); o.y = pack2(a.z, a.w);
  o.z = pack2(b.x, b.y); o.w = pack2(b.z, b.w);
  *(uint4*)(kwb + i) = o;
  a = *(const float4*)(vw + i); b = *(const float4*)(vw + i + 4);
  o.x = pack2(a.x, a.y); o.y = pack2(a.z, a.w);
  o.z = pack2(b.x, b.y); o.w = pack2(b.z, b.w);
  *(uint4*)(vwb + i) = o;
}

// ------------------------------------------------ TDM descriptor helpers
// D# group 0: count=1, lds_addr, global_addr, type=2 ("image").
__device__ __forceinline__ v4u tdm_g0(unsigned lds, unsigned long long ga) {
  v4u g;
  g.x = 1u;                                    // count=1, user descriptor
  g.y = lds;                                   // LDS byte address
  g.z = (unsigned)(ga & 0xFFFFFFFFu);          // global_addr[31:0]
  g.w = (unsigned)((ga >> 32) & 0x01FFFFFFu)   // global_addr[56:32]
        | (2u << 30);                          // type=2
  return g;
}
// D# group 1: data_size=1 (2B), LDS pad: every 16 DWORDs insert 8 DWORDs
// (row 32 bf16 = 64B -> LDS row stride 96B = 48 elements), 2-D tile.
__device__ __forceinline__ v8i tdm_g1(unsigned td0, unsigned td1,
                                      unsigned tl0, unsigned tl1,
                                      unsigned long long s0) {
  v8i g;
  g[0] = (int)((1u << 16) | (1u << 20) | (3u << 22) | (7u << 25));
  g[1] = (int)((td0 & 0xFFFFu) << 16);                       // tensor_dim0 lo
  g[2] = (int)((td0 >> 16) | ((td1 & 0xFFFFu) << 16));       // td0 hi | td1 lo
  g[3] = (int)((td1 >> 16) | (tl0 << 16));                   // td1 hi | tile_dim0
  g[4] = (int)tl1;                                           // tile_dim1, tile_dim2=0
  g[5] = (int)(s0 & 0xFFFFFFFFull);                          // dim0 stride lo
  g[6] = (int)((s0 >> 32) & 0xFFFFull);                      // stride hi | stride1 lo = 0
  g[7] = 0;
  return g;
}
__device__ __forceinline__ unsigned lds_addr32(const void* p) {
  return (unsigned)(uintptr_t)p;  // LDS generic addr: [31:0] = LDS byte offset
}

// ---------------------------------------------------------------- kernel 3
// Dual GEMM (K and V projections) on pre-normalized bf16 activations.
// Block tile 64(M) x 128(N), 8 waves, wave tile 32x32 per output matrix.
// Double-buffered LDS filled by the Tensor Data Mover (TENSORcnt), compute
// is pure ds_load_b128 + v_wmma_f32_16x16x32_bf16.
__global__ void __launch_bounds__(256, 2)
k_dual_gemm(const unsigned short* __restrict__ hsb,
            const unsigned short* __restrict__ kwb,
            const unsigned short* __restrict__ vwb,
            const float* __restrict__ kbias, const float* __restrict__ vbias,
            float* __restrict__ kout, float* __restrict__ vout) {
  __shared__ unsigned short As[2][64 * 48];     // row stride 48 (TDM pad)
  __shared__ unsigned short Bk[2][128 * 48];
  __shared__ unsigned short Bv[2][128 * 48];

  const int tid  = threadIdx.x;
  const int wave = tid >> 5, lane = tid & 31;
  const int hi   = lane >> 4, l15 = lane & 15;
  const int wm   = wave & 1,  wn  = wave >> 1;
  const int n0   = blockIdx.x * 128;
  const int m0   = blockIdx.y * 64;

  v8f ak[2][2] = {};
  v8f av[2][2] = {};

#if __has_builtin(__builtin_amdgcn_tensor_load_to_lds)
  const v8i g1A = tdm_g1(E_, B_ * S_, 32, 64, E_);    // A: 64 rows x 32 K
  const v8i g1B = tdm_g1(E_, E_, 32, 128, E_);        // B: 128 rows x 32 K
  const v4i gz4 = {0, 0, 0, 0};
  const v8i gz8 = {0, 0, 0, 0, 0, 0, 0, 0};
  const bool issuer = (wave == 0);

  #define TDM_ISSUE(buf_, k0_)                                                  \
    do {                                                                        \
      unsigned long long gaA =                                                  \
          (unsigned long long)(uintptr_t)(hsb + (size_t)m0 * E_ + (k0_));       \
      unsigned long long gaK =                                                  \
          (unsigned long long)(uintptr_t)(kwb + (size_t)n0 * E_ + (k0_));       \
      unsigned long long gaV =                                                  \
          (unsigned long long)(uintptr_t)(vwb + (size_t)n0 * E_ + (k0_));       \
      __builtin_amdgcn_tensor_load_to_lds(tdm_g0(lds_addr32(&As[buf_][0]), gaA),\
                                          g1A, gz4, gz4, gz8, 0);               \
      __builtin_amdgcn_tensor_load_to_lds(tdm_g0(lds_addr32(&Bk[buf_][0]), gaK),\
                                          g1B, gz4, gz4, gz8, 0);               \
      __builtin_amdgcn_tensor_load_to_lds(tdm_g0(lds_addr32(&Bv[buf_][0]), gaV),\
                                          g1B, gz4, gz4, gz8, 0);               \
    } while (0)

  if (issuer) TDM_ISSUE(0, 0);
#endif

  int buf = 0;
  for (int k0 = 0; k0 < E_; k0 += 32) {
#if __has_builtin(__builtin_amdgcn_tensor_load_to_lds)
    if (issuer) __builtin_amdgcn_s_wait_tensorcnt(0);   // buf tiles landed
    __syncthreads();                                    // all done with buf^1
    if (issuer && (k0 + 32 < E_)) TDM_ISSUE(buf ^ 1, k0 + 32);
#else
    // Fallback: manual bf16 global->LDS copy (16B per thread per slot).
    __syncthreads();
    {
      int r = tid >> 2, c8 = (tid & 3) * 8;             // A: 64x32, 256 slots
      *(uint4*)(&As[buf][r * 48 + c8]) =
          *(const uint4*)(hsb + (size_t)(m0 + r) * E_ + k0 + c8);
#pragma unroll
      for (int rep = 0; rep < 2; ++rep) {               // B: 128x32, 512 slots
        int slot = tid + rep * 256;
        int rb = slot >> 2, cb = (slot & 3) * 8;
        *(uint4*)(&Bk[buf][rb * 48 + cb]) =
            *(const uint4*)(kwb + (size_t)(n0 + rb) * E_ + k0 + cb);
        *(uint4*)(&Bv[buf][rb * 48 + cb]) =
            *(const uint4*)(vwb + (size_t)(n0 + rb) * E_ + k0 + cb);
      }
    }
    __syncthreads();
#endif

    // ---- fragments (documented 16-bit WMMA layouts)
    v16bf a[2], bk[2], bv[2];
#pragma unroll
    for (int i = 0; i < 2; ++i) {            // A: lane row = l15, K split by hi
      int m = wm * 32 + 16 * i + l15;
      FragU f;
      f.q[0] = *(const uint4*)(&As[buf][m * 48 + 8 * hi]);      // K = 8*hi+0..7
      f.q[1] = *(const uint4*)(&As[buf][m * 48 + 16 + 8 * hi]); // K = 16+8*hi+..
      a[i] = f.v;
    }
#pragma unroll
    for (int j = 0; j < 2; ++j) {            // B: lane col = l15, K split by hi (same as A)
      int n = wn * 32 + 16 * j + l15;
      FragU fk, fv;
      fk.q[0] = *(const uint4*)(&Bk[buf][n * 48 + 8 * hi]);
      fk.q[1] = *(const uint4*)(&Bk[buf][n * 48 + 16 + 8 * hi]);
      bk[j] = fk.v;
      fv.q[0] = *(const uint4*)(&Bv[buf][n * 48 + 8 * hi]);
      fv.q[1] = *(const uint4*)(&Bv[buf][n * 48 + 16 + 8 * hi]);
      bv[j] = fv.v;
    }
#pragma unroll
    for (int i = 0; i < 2; ++i)
#pragma unroll
      for (int j = 0; j < 2; ++j) {
        ak[i][j] = __builtin_amdgcn_wmma_f32_16x16x32_bf16(
            false, a[i], false, bk[j], (short)0, ak[i][j], false, false);
        av[i][j] = __builtin_amdgcn_wmma_f32_16x16x32_bf16(
            false, a[i], false, bv[j], (short)0, av[i][j], false, false);
      }

    buf ^= 1;
  }

  // ---- epilogue: bias + store (C/D layout: col = l15, row = 8*hi + r)
#pragma unroll
  for (int j = 0; j < 2; ++j) {
    int gn = n0 + wn * 32 + 16 * j + l15;
    float kbv = kbias[gn], vbv = vbias[gn];
#pragma unroll
    for (int i = 0; i < 2; ++i) {
      int gmB = m0 + wm * 32 + 16 * i + 8 * hi;
#pragma unroll
      for (int r = 0; r < 8; ++r) {
        size_t o = (size_t)(gmB + r) * E_ + gn;
        kout[o] = ak[i][j][r] + kbv;
        vout[o] = av[i][j][r] + vbv;
      }
    }
  }
}

// ---------------------------------------------------------------- kernel 4
// q[b,:] = hs_bf16[b, ix[b], :] @ q_w^T + q_b   (tiny: 8 rows)
__global__ void k_q_proj(const unsigned short* __restrict__ hsb,
                         const float* __restrict__ qw, const float* __restrict__ qbias,
                         const int* __restrict__ ix, float* __restrict__ q) {
  int b = blockIdx.y, tid = threadIdx.x;
  int e = blockIdx.x * 256 + tid;
  int row = b * S_ + ix[b];
  const unsigned short* hr = hsb + (size_t)row * E_;
  const float* wr = qw + (size_t)e * E_;
  __shared__ float h_s[256];
  float acc = 0.f;
  for (int jc = 0; jc < E_; jc += 256) {
    __syncthreads();
    h_s[tid] = bf2f(hr[jc + tid]);
    __syncthreads();
    for (int j = 0; j < 256; ++j) acc += h_s[j] * wr[jc + j];
  }
  q[b * E_ + e] = acc + qbias[e];
}

// ---------------------------------------------------------------- kernel 5
// In-place RoPE on q (first 64 dims of each head, pos = ix[b]).
__global__ void k_rope_q(float* __restrict__ q, const int* __restrict__ ix) {
  int b = blockIdx.x, tid = threadIdx.x;
  float pos = (float)ix[b];
#pragma unroll
  for (int rep = 0; rep < 2; ++rep) {
    int p = tid + rep * 256;
    int h = p >> 5, i = p & 31;
    float invf = __expf(-(float)(2 * i) * 0.14391157f);  // ln(10000)/64
    float sn, cs;
    __sincosf(pos * invf, &sn, &cs);
    float* qp = q + b * E_ + h * D_ + 2 * i;
    float x1 = qp[0], x2 = qp[1];
    qp[0] = x1 * cs - x2 * sn;
    qp[1] = x2 * cs + x1 * sn;
  }
}

// ---------------------------------------------------------------- kernel 6
// One block per (b,h): RoPE(k) on the fly, unscaled scores (matches ref),
// softmax, weighted sum over V.
__global__ void k_attention(const float* __restrict__ qb, const float* __restrict__ kb,
                            const float* __restrict__ vb, float* __restrict__ ao) {
  int bh = blockIdx.x, tid = threadIdx.x;
  int b = bh >> 4, h = bh & 15;
  __shared__ float qs[D_];
  __shared__ float invfs[32];
  __shared__ float p[S_];
  __shared__ float red[256];
  __shared__ float sZ;
  if (tid < D_) qs[tid] = qb[b * E_ + h * D_ + tid];
  if (tid < 32) invfs[tid] = __expf(-(float)(2 * tid) * 0.14391157f);
  __syncthreads();

  float lmax = -3.4e38f;
  float sc[8];
  for (int u = 0; u < 8; ++u) {
    int s = tid + u * 256;
    const float* kp = kb + ((size_t)(b * S_ + s)) * E_ + h * D_;
    float acc = 0.f;
    float fs = (float)s;
#pragma unroll
    for (int i = 0; i < 32; ++i) {
      float sn, cs;
      __sincosf(fs * invfs[i], &sn, &cs);
      float k0 = kp[2 * i], k1 = kp[2 * i + 1];
      acc += qs[2 * i]     * (k0 * cs - k1 * sn)
           + qs[2 * i + 1] * (k1 * cs + k0 * sn);
    }
#pragma unroll
    for (int d2 = 64; d2 < 128; ++d2) acc += qs[d2] * kp[d2];
    sc[u] = acc;
    lmax = fmaxf(lmax, acc);
  }
  red[tid] = lmax; __syncthreads();
  for (int off = 128; off > 0; off >>= 1) {
    if (tid < off) red[tid] = fmaxf(red[tid], red[tid + off]);
    __syncthreads();
  }
  float mx = red[0];
  __syncthreads();
  float lsum = 0.f;
  for (int u = 0; u < 8; ++u) {
    float e = __expf(sc[u] - mx);
    p[tid + u * 256] = e;
    lsum += e;
  }
  red[tid] = lsum; __syncthreads();
  for (int off = 128; off > 0; off >>= 1) {
    if (tid < off) red[tid] += red[tid + off];
    __syncthreads();
  }
  if (tid == 0) sZ = red[0];
  __syncthreads();
  float invZ = 1.f / sZ;

  int d = tid & 127, half = tid >> 7;
  float acc = 0.f;
  for (int s = half; s < S_; s += 2)
    acc += p[s] * vb[((size_t)(b * S_ + s)) * E_ + h * D_ + d];
  red[tid] = acc; __syncthreads();
  if (tid < 128)
    ao[b * E_ + h * D_ + tid] = (red[tid] + red[tid + 128]) * invZ;
}

// ---------------------------------------------------------------- kernel 7
// out[b,e] = ao[b,:] @ out_w[e,:] + out_b[e]  (8x2048, LDS-staged activations)
__global__ void k_out_proj(const float* __restrict__ ao, const float* __restrict__ ow,
                           const float* __restrict__ ob, float* __restrict__ out) {
  int tid = threadIdx.x;
  int e = blockIdx.x * 256 + tid;
  __shared__ float a_s[8 * 256];
  float acc[B_] = {};
  const float* wr = ow + (size_t)e * E_;
  for (int jc = 0; jc < E_; jc += 256) {
    __syncthreads();
#pragma unroll
    for (int r = 0; r < 8; ++r) {
      int idx = r * 256 + tid;
      a_s[idx] = ao[(idx >> 8) * E_ + jc + (idx & 255)];
    }
    __syncthreads();
    for (int j = 0; j < 256; ++j) {
      float w = wr[jc + j];
#pragma unroll
      for (int bb = 0; bb < B_; ++bb) acc[bb] += a_s[bb * 256 + j] * w;
    }
  }
  float bias = ob[e];
#pragma unroll
  for (int bb = 0; bb < B_; ++bb) out[bb * E_ + e] = acc[bb] + bias;
}

// ---------------------------------------------------------------- launch
extern "C" void kernel_launch(void* const* d_in, const int* in_sizes, int n_in,
                              void* d_out, int out_size, void* d_ws, size_t ws_size,
                              hipStream_t stream) {
  (void)in_sizes; (void)n_in; (void)out_size; (void)ws_size;
  const float* hidden = (const float*)d_in[0];
  const int*   ids    = (const int*)  d_in[1];
  const float* lnw    = (const float*)d_in[2];
  const float* lnb    = (const float*)d_in[3];
  const float* qw     = (const float*)d_in[4];
  const float* qb     = (const float*)d_in[5];
  const float* kw     = (const float*)d_in[6];
  const float* kbias  = (const float*)d_in[7];
  const float* vw     = (const float*)d_in[8];
  const float* vbias  = (const float*)d_in[9];
  const float* ow     = (const float*)d_in[10];
  const float* ob     = (const float*)d_in[11];

  char* ws = (char*)d_ws;
  int*            ixp  = (int*)ws;                               // 32 B
  unsigned short* hsb  = (unsigned short*)(ws + 256);            // 67108864 B
  unsigned short* kwb  = (unsigned short*)(ws + 67109120);       // 8388608 B
  unsigned short* vwb  = (unsigned short*)(ws + 75497728);       // 8388608 B
  float*          qbuf = (float*)(ws + 83886336);                // 65536 B
  float*          aob  = (float*)(ws + 83951872);                // 65536 B
  float*          kbuf = (float*)(ws + 84017408);                // 134217728 B
  float*          vbuf = (float*)(ws + 218235136);               // 134217728 B

  k_find_ix  <<<B_, 256, 0, stream>>>(ids, ixp);
  k_ln_bf16  <<<B_ * S_, 256, 0, stream>>>(hidden, lnw, lnb, hsb);
  k_w2bf     <<<(E_ * E_) / (256 * 8), 256, 0, stream>>>(kw, vw, kwb, vwb);
  k_dual_gemm<<<dim3(E_ / 128, (B_ * S_) / 64), 256, 0, stream>>>(
      hsb, kwb, vwb, kbias, vbias, kbuf, vbuf);
  k_q_proj   <<<dim3(E_ / 256, B_), 256, 0, stream>>>(hsb, qw, qb, ixp, qbuf);
  k_rope_q   <<<B_, 256, 0, stream>>>(qbuf, ixp);
  k_attention<<<B_ * H_, 256, 0, stream>>>(qbuf, kbuf, vbuf, aob);
  k_out_proj <<<E_ / 256, 256, 0, stream>>>(aob, ow, ob, (float*)d_out);
}
